// EvolvedMultiHeadAttention_53412213293573
// MI455X (gfx1250) — matrix operand order repaired and checked by reading, not compiled, not run on hardware
//
#include <hip/hip_runtime.h>
#include <hip/hip_bf16.h>
#include <math.h>

// MI455X / gfx1250, wave32. bf16 WMMA (v_wmma_f32_16x16x32_bf16) path +
// GLOBAL_LOAD_ASYNC_TO_LDS double-buffered staging in the attention kernel.

typedef __attribute__((ext_vector_type(16))) __bf16 v16bf;
typedef __attribute__((ext_vector_type(8)))  __bf16 v8bf;
typedef __attribute__((ext_vector_type(8)))  float  v8f;

#define WMMA_BF16(A, B, C) \
  __builtin_amdgcn_wmma_f32_16x16x32_bf16(false, (A), false, (B), (short)0, (C), false, false)

static __device__ __forceinline__ v16bf cat16(v8bf lo, v8bf hi) {
  return __builtin_shufflevector(lo, hi, 0,1,2,3,4,5,6,7,8,9,10,11,12,13,14,15);
}

// A-matrix (16x32 bf16) fragment. p = row base (+ kstep offset). half = lane>>4.
// lane<16: K 0..7 and 16..23 ; lane>=16: K 8..15 and 24..31  (ISA 7.12.2)
static __device__ __forceinline__ v16bf load_a_frag(const __bf16* p, int half) {
  const int koff = half * 8;
  v8bf lo = *(const v8bf*)(p + koff);
  v8bf hi = *(const v8bf*)(p + koff + 16);
  return cat16(lo, hi);
}

// B-matrix (32x16 bf16) fragment loaded from B^T rows (row = output col n).
// lane<16: K 0..7 and 16..23 ; lane>=16: K 8..15 and 24..31
static __device__ __forceinline__ v16bf load_b_frag(const __bf16* p, int half) {
  const int koff = half * 8;
  v8bf lo = *(const v8bf*)(p + koff);
  v8bf hi = *(const v8bf*)(p + koff + 16);
  return cat16(lo, hi);
}

static __device__ __forceinline__ v8f vzero8() {
  v8f z = {0.f,0.f,0.f,0.f,0.f,0.f,0.f,0.f};
  return z;
}

// ---- CDNA5 async Global->LDS (tracked by ASYNCcnt), ISA ch.10 / 15.18.3 ----
// VDST vgpr = LDS byte offset (low 32 bits of generic shared pointer), GV mode.
static __device__ __forceinline__ unsigned lds_off(const void* p) {
  return (unsigned)(size_t)p;  // generic shared addr = {aperture_hi, lds_offset}
}
static __device__ __forceinline__ void async_ld_b128(unsigned ldsoff, const void* g) {
  asm volatile("global_load_async_to_lds_b128 %0, %1, off"
               :: "v"(ldsoff), "v"(g) : "memory");
}
#define S_WAIT_ASYNCCNT(n) asm volatile("s_wait_asynccnt " #n ::: "memory")

// ---------------------------------------------------------------------------
// Kernel 1: fused QKV projection. y = x @ W^T + b, cast to bf16.
// mode = blockIdx.z : 0 -> Q [b,h,S,hd], 1 -> K [b,h,S,hd], 2 -> V^T [b,h,hd,S]
// M = B*S = 4096, N = D = 1024, K = 1024. Block tile 128x128, BK = 32.
// ---------------------------------------------------------------------------
__global__ __launch_bounds__(256) void proj_gemm_kernel(
    const float* __restrict__ X,
    const float* __restrict__ Wq, const float* __restrict__ bq,
    const float* __restrict__ Wk, const float* __restrict__ bk,
    const float* __restrict__ Wv, const float* __restrict__ bv,
    __bf16* __restrict__ Qws, __bf16* __restrict__ Kws, __bf16* __restrict__ Vtws)
{
  const int mode = blockIdx.z;
  const float* W    = (mode == 0) ? Wq : (mode == 1) ? Wk : Wv;
  const float* bias = (mode == 0) ? bq : (mode == 1) ? bk : bv;
  __bf16* dst       = (mode == 0) ? Qws : (mode == 1) ? Kws : Vtws;

  __shared__ __bf16 As[128 * 40];   // 128 rows x 32 k, pad to 40 (80B = 16B*5)
  __shared__ __bf16 Bs[128 * 40];

  const int tid  = threadIdx.x;
  const int lane = tid & 31;
  const int wave = tid >> 5;
  const int wm   = wave & 3;    // 4 waves along M (32 rows each)
  const int wn   = wave >> 2;   // 2 waves along N (64 cols each)
  const int half = lane >> 4;
  const int l16  = lane & 15;
  const int m0   = blockIdx.x * 128;
  const int n0   = blockIdx.y * 128;

  v8f acc[2][4];
#pragma unroll
  for (int i = 0; i < 2; ++i)
#pragma unroll
    for (int j = 0; j < 4; ++j) acc[i][j] = vzero8();

  for (int kb = 0; kb < 1024; kb += 32) {
    __syncthreads();
    // stage A (x) and B (W) tiles, converting f32 -> bf16
#pragma unroll
    for (int i = 0; i < 4; ++i) {
      const int c   = tid + i * 256;       // float4 chunk 0..1023
      const int row = c >> 3;
      const int col = (c & 7) << 2;
      const float* pax = X + (size_t)(m0 + row) * 1024 + kb + col;
      const float* pbx = W + (size_t)(n0 + row) * 1024 + kb + col;
      if (kb + 32 < 1024) {               // global_prefetch_b8 next k-slab
        __builtin_prefetch(pax + 32, 0, 1);
        __builtin_prefetch(pbx + 32, 0, 1);
      }
      const float4 va = *(const float4*)pax;
      __bf16* pa = As + row * 40 + col;
      pa[0] = (__bf16)va.x; pa[1] = (__bf16)va.y; pa[2] = (__bf16)va.z; pa[3] = (__bf16)va.w;
      const float4 vb = *(const float4*)pbx;
      __bf16* pb = Bs + row * 40 + col;
      pb[0] = (__bf16)vb.x; pb[1] = (__bf16)vb.y; pb[2] = (__bf16)vb.z; pb[3] = (__bf16)vb.w;
    }
    __syncthreads();

    v16bf af[2], bfr[4];
#pragma unroll
    for (int i = 0; i < 2; ++i)
      af[i] = load_a_frag(As + (wm * 32 + i * 16 + l16) * 40, half);
#pragma unroll
    for (int j = 0; j < 4; ++j)
      bfr[j] = load_b_frag(Bs + (wn * 64 + j * 16 + l16) * 40, half);
#pragma unroll
    for (int i = 0; i < 2; ++i)
#pragma unroll
      for (int j = 0; j < 4; ++j)
        acc[i][j] = WMMA_BF16(af[i], bfr[j], acc[i][j]);
  }

  // epilogue: bias + cast + scatter to per-head layout
#pragma unroll
  for (int i = 0; i < 2; ++i)
#pragma unroll
    for (int j = 0; j < 4; ++j)
#pragma unroll
      for (int p = 0; p < 8; ++p) {
        const int m = m0 + wm * 32 + i * 16 + p + half * 8;
        const int n = n0 + wn * 64 + j * 16 + l16;
        const float v = acc[i][j][p] + bias[n];
        const int b = m >> 11, s = m & 2047;    // S = 2048
        const int h = n >> 6,  d = n & 63;      // hd = 64
        size_t o;
        if (mode != 2) o = (((size_t)(b * 16 + h) * 2048 + s) * 64 + d);
        else           o = (((size_t)(b * 16 + h) * 64 + d) * 2048 + s);
        dst[o] = (__bf16)v;
      }
}

// ---------------------------------------------------------------------------
// Kernel 2: flash attention. grid (S/64, B*H), 128 thr = 4 waves x 16 q rows.
// Double-buffered GLOBAL_LOAD_ASYNC_TO_LDS staging of K / V^T / mask tiles.
// ---------------------------------------------------------------------------
__global__ __launch_bounds__(128) void attn_kernel(
    const __bf16* __restrict__ Qws, const __bf16* __restrict__ Kws,
    const __bf16* __restrict__ Vtws, const unsigned char* __restrict__ mask,
    __bf16* __restrict__ ctx)
{
  const int bh   = blockIdx.y;          // b*16 + h
  const int q0   = blockIdx.x * 64;
  const int tid  = threadIdx.x;
  const int lane = tid & 31;
  const int wave = tid >> 5;
  const int half = lane >> 4;
  const int l16  = lane & 15;

  __shared__ __bf16 Ks[2][64 * 72];        // [key][hd]   (pad 72: 144B = 16B*9)
  __shared__ __bf16 Vs[2][64 * 72];        // [hd][key]   (V^T tile)
  __shared__ unsigned char Ms[2][64 * 64]; // [q][key] keep bytes
  __shared__ __bf16 Ps[4][16 * 72];        // per-wave P C-layout -> A-layout bounce

  const unsigned char* mrow = mask + ((size_t)bh * 2048 + q0) * 2048;

  // issue one tile's async copies: 4x K + 4x V^T + 2x mask b128 per thread = 10
  auto issue = [&](int kb, int buf) {
#pragma unroll
    for (int i = 0; i < 4; ++i) {
      const int c   = tid + i * 128;   // 16B chunk 0..511
      const int r   = c >> 3;
      const int col = (c & 7) * 8;     // bf16 units
      async_ld_b128(lds_off(&Ks[buf][r * 72 + col]),
                    Kws + ((size_t)bh * 2048 + kb + r) * 64 + col);
      async_ld_b128(lds_off(&Vs[buf][r * 72 + col]),
                    Vtws + ((size_t)bh * 64 + r) * 2048 + kb + col);
    }
#pragma unroll
    for (int i = 0; i < 2; ++i) {
      const int c   = tid + i * 128;   // 16B chunk 0..255
      const int r   = c >> 2;
      const int col = (c & 3) * 16;
      async_ld_b128(lds_off(&Ms[buf][r * 64 + col]),
                    mrow + (size_t)r * 2048 + kb + col);
    }
  };

  // Q fragments for this wave's 16 rows, 2 k-steps over hd=64
  v16bf qf[2];
  {
    const __bf16* qp = Qws + ((size_t)bh * 2048 + q0 + wave * 16 + l16) * 64;
    qf[0] = load_a_frag(qp, half);
    qf[1] = load_a_frag(qp + 32, half);
  }

  float rmax[8], rsum[8];
  v8f cacc[4];
#pragma unroll
  for (int p = 0; p < 8; ++p) { rmax[p] = -INFINITY; rsum[p] = 0.f; }
#pragma unroll
  for (int t = 0; t < 4; ++t) cacc[t] = vzero8();

  const float scale = 0.125f;  // hd^-0.5 = 1/8

  issue(0, 0);                 // prologue: tile 0 in flight

  for (int it = 0; it < 32; ++it) {
    const int buf = it & 1;
    if (it + 1 < 32) {
      issue((it + 1) * 64, buf ^ 1);   // buf^1 last read in iter it-1 (barrier'd)
      S_WAIT_ASYNCCNT(10);             // in-order: batch `it` landed, `it+1` in flight
    } else {
      S_WAIT_ASYNCCNT(0);
    }
    __syncthreads();

    // scores S = Q @ K^T  (4 n-tiles of 16x16, 2 k-steps)
    v8f sacc[4];
#pragma unroll
    for (int t = 0; t < 4; ++t) {
      sacc[t] = vzero8();
      const __bf16* kp = &Ks[buf][(t * 16 + l16) * 72];
      sacc[t] = WMMA_BF16(qf[0], load_b_frag(kp, half), sacc[t]);
      sacc[t] = WMMA_BF16(qf[1], load_b_frag(kp + 32, half), sacc[t]);
    }

    // online softmax in C-layout: row m = p + 8*half, col = t*16 + l16
#pragma unroll
    for (int p = 0; p < 8; ++p) {
      const int m    = p + half * 8;
      const int qrow = wave * 16 + m;
      float s[4];
      bool  keep[4];
      float mx = -INFINITY;
#pragma unroll
      for (int t = 0; t < 4; ++t) {
        keep[t] = Ms[buf][qrow * 64 + t * 16 + l16] != 0;
        s[t] = sacc[t][p] * scale;
        if (keep[t]) mx = fmaxf(mx, s[t]);
      }
#pragma unroll
      for (int d = 1; d < 16; d <<= 1) mx = fmaxf(mx, __shfl_xor(mx, d, 32));
      const float nm   = fmaxf(rmax[p], mx);
      const float corr = (nm == -INFINITY) ? 1.0f : __expf(rmax[p] - nm);
      float psum = 0.f;
#pragma unroll
      for (int t = 0; t < 4; ++t) {
        const float pv = keep[t] ? __expf(s[t] - nm) : 0.f;
        psum += pv;
        Ps[wave][m * 72 + t * 16 + l16] = (__bf16)pv;
      }
#pragma unroll
      for (int d = 1; d < 16; d <<= 1) psum += __shfl_xor(psum, d, 32);
      rsum[p] = rsum[p] * corr + psum;
      rmax[p] = nm;
#pragma unroll
      for (int tc = 0; tc < 4; ++tc) cacc[tc][p] *= corr;
    }

    // ctx += P @ V   (P from per-wave LDS in A-layout; V^T rows from Vs)
#pragma unroll
    for (int ks = 0; ks < 2; ++ks) {
      v16bf pf = load_a_frag(&Ps[wave][l16 * 72 + ks * 32], half);
#pragma unroll
      for (int tc = 0; tc < 4; ++tc) {
        v16bf vf = load_b_frag(&Vs[buf][(tc * 16 + l16) * 72 + ks * 32], half);
        cacc[tc] = WMMA_BF16(pf, vf, cacc[tc]);
      }
    }
    __syncthreads();   // all waves done with buf before batch it+2 overwrites it
  }

  // epilogue: normalize (fully-masked rows -> 0), write ctx [B,S,D] bf16
  const int b = bh >> 4, h = bh & 15;
#pragma unroll
  for (int tc = 0; tc < 4; ++tc)
#pragma unroll
    for (int p = 0; p < 8; ++p) {
      const int m = p + half * 8;
      const int q = q0 + wave * 16 + m;
      const float l = rsum[p];
      const float v = (l > 0.f) ? cacc[tc][p] / l : 0.f;
      ctx[(size_t)(b * 2048 + q) * 1024 + h * 64 + tc * 16 + l16] = (__bf16)v;
    }
}

// ---------------------------------------------------------------------------
// Kernel 3: out = ctx @ Wo^T + bo (f32 output). Same tiling as kernel 1.
// ---------------------------------------------------------------------------
__global__ __launch_bounds__(256) void outproj_kernel(
    const __bf16* __restrict__ Ctx, const float* __restrict__ Wo,
    const float* __restrict__ bo, float* __restrict__ out)
{
  __shared__ __bf16 As[128 * 40];
  __shared__ __bf16 Bs[128 * 40];

  const int tid  = threadIdx.x;
  const int lane = tid & 31;
  const int wave = tid >> 5;
  const int wm   = wave & 3;
  const int wn   = wave >> 2;
  const int half = lane >> 4;
  const int l16  = lane & 15;
  const int m0   = blockIdx.x * 128;
  const int n0   = blockIdx.y * 128;

  v8f acc[2][4];
#pragma unroll
  for (int i = 0; i < 2; ++i)
#pragma unroll
    for (int j = 0; j < 4; ++j) acc[i][j] = vzero8();

  for (int kb = 0; kb < 1024; kb += 32) {
    __syncthreads();
    // A tile: already bf16, 512 x 16B chunks
#pragma unroll
    for (int i = 0; i < 2; ++i) {
      const int c   = tid + i * 256;
      const int r   = c >> 2;
      const int col = (c & 3) * 8;
      const __bf16* pax = Ctx + (size_t)(m0 + r) * 1024 + kb + col;
      if (kb + 32 < 1024) __builtin_prefetch(pax + 32, 0, 1);
      *(uint4*)(As + r * 40 + col) = *(const uint4*)pax;
    }
    // B tile: Wo f32 -> bf16
#pragma unroll
    for (int i = 0; i < 4; ++i) {
      const int c   = tid + i * 256;
      const int row = c >> 3;
      const int col = (c & 7) << 2;
      const float* pbx = Wo + (size_t)(n0 + row) * 1024 + kb + col;
      if (kb + 32 < 1024) __builtin_prefetch(pbx + 32, 0, 1);
      const float4 vb = *(const float4*)pbx;
      __bf16* pb = Bs + row * 40 + col;
      pb[0] = (__bf16)vb.x; pb[1] = (__bf16)vb.y; pb[2] = (__bf16)vb.z; pb[3] = (__bf16)vb.w;
    }
    __syncthreads();

    v16bf af[2], bfr[4];
#pragma unroll
    for (int i = 0; i < 2; ++i)
      af[i] = load_a_frag(As + (wm * 32 + i * 16 + l16) * 40, half);
#pragma unroll
    for (int j = 0; j < 4; ++j)
      bfr[j] = load_b_frag(Bs + (wn * 64 + j * 16 + l16) * 40, half);
#pragma unroll
    for (int i = 0; i < 2; ++i)
#pragma unroll
      for (int j = 0; j < 4; ++j)
        acc[i][j] = WMMA_BF16(af[i], bfr[j], acc[i][j]);
  }

#pragma unroll
  for (int i = 0; i < 2; ++i)
#pragma unroll
    for (int j = 0; j < 4; ++j)
#pragma unroll
      for (int p = 0; p < 8; ++p) {
        const int m = m0 + wm * 32 + i * 16 + p + half * 8;
        const int n = n0 + wn * 64 + j * 16 + l16;
        out[(size_t)m * 1024 + n] = acc[i][j][p] + bo[n];
      }
}

// ---------------------------------------------------------------------------
extern "C" void kernel_launch(void* const* d_in, const int* in_sizes, int n_in,
                              void* d_out, int out_size, void* d_ws, size_t ws_size,
                              hipStream_t stream) {
  (void)in_sizes; (void)n_in; (void)out_size; (void)ws_size;
  const float* x            = (const float*)d_in[0];
  const unsigned char* mask = (const unsigned char*)d_in[1];  // jnp bool = 1 byte
  const float* Wq = (const float*)d_in[2];
  const float* bq = (const float*)d_in[3];
  const float* Wk = (const float*)d_in[4];
  const float* bk = (const float*)d_in[5];
  const float* Wv = (const float*)d_in[6];
  const float* bv = (const float*)d_in[7];
  const float* Wo = (const float*)d_in[8];
  const float* bo = (const float*)d_in[9];
  float* out = (float*)d_out;

  // workspace: Q, K, V^T, ctx in bf16 (4 x 8MB = 32MB)
  const size_t n1 = (size_t)2 * 16 * 2048 * 64;
  __bf16* Qws = (__bf16*)d_ws;
  __bf16* Kws = Qws + n1;
  __bf16* Vt  = Kws + n1;
  __bf16* Ctx = Vt + n1;

  proj_gemm_kernel<<<dim3(32, 8, 3), 256, 0, stream>>>(
      x, Wq, bq, Wk, bk, Wv, bv, Qws, Kws, Vt);
  attn_kernel<<<dim3(32, 32), 128, 0, stream>>>(Qws, Kws, Vt, mask, Ctx);
  outproj_kernel<<<dim3(32, 8), 256, 0, stream>>>(Ctx, Wo, bo, out);
}
